// EdgeCentricGNN_32487132627075
// MI455X (gfx1250) — compile-verified, matrix operand order checked by reading, not run on hardware
//
#include <hip/hip_runtime.h>

// =====================================================================
// EdgeCentricGNN forward for MI455X (gfx1250, wave32, WMMA).
//
// All dense linears run through v_wmma_f32_16x16x32_f16 (f16 weights,
// f32 accumulate).  Edge feature tensor `e` is stored f16 (GEMM-input
// only).  Scatter/softmax parts are bandwidth-bound elementwise kernels
// with hardware f32 atomics.
// =====================================================================

#define NNODES 50000
#define NEDGES 600000
#define NGRAPH 128
#define HID    128

typedef __attribute__((ext_vector_type(16))) _Float16 v16h;
typedef __attribute__((ext_vector_type(8)))  _Float16 v8h;
typedef __attribute__((ext_vector_type(8)))  float    v8f;

// ---------------------------------------------------------------------
// Input pointer indices.
// Assumption: d_in[] = top-level setup_inputs() dict insertion order
// (x, edge_index, edge_attr, batch), then the params pytree flattened in
// JAX order (dict keys sorted alphabetically, recursively; lists in
// order; each _lin contributes "b" then "w").
// ---------------------------------------------------------------------
enum ParamIdx {
  IN_X = 0, IN_EIDX = 1, IN_EATTR = 2, IN_BATCH = 3,
  BN2A_B = 4, BN2A_G, BN2B_B, BN2B_G, BN2BLK_B, BN2BLK_G,
  BN3A_B, BN3A_G, BN3B_B, BN3B_G, BN3BLK_B, BN3BLK_G,
  BNFIN_B, BNFIN_G, BNMID_B, BNMID_G, BNMID2_B, BNMID2_G, BNMID3_B, BNMID3_G,
  C2A_EPS, C2A_LIN_B, C2A_LIN_W, C2A_NN1_B, C2A_NN1_W, C2A_NN2_B, C2A_NN2_W,
  C2B_EPS, C2B_LIN_B, C2B_LIN_W, C2B_NN1_B, C2B_NN1_W, C2B_NN2_B, C2B_NN2_W,
  C3A_EPS, C3A_LIN_B, C3A_LIN_W, C3A_NN1_B, C3A_NN1_W, C3A_NN2_B, C3A_NN2_W,
  C3B_EPS, C3B_LIN_B, C3B_LIN_W, C3B_NN1_B, C3B_NN1_W, C3B_NN2_B, C3B_NN2_W,
  CF_E_W, CF_K_B, CF_K_W, CF_Q_B, CF_Q_W, CF_SK_B, CF_SK_W, CF_V_B, CF_V_W,
  CM_E_W, CM_K_B, CM_K_W, CM_Q_B, CM_Q_W, CM_SK_B, CM_SK_W, CM_V_B, CM_V_W,
  CM2_E_W, CM2_K_B, CM2_K_W, CM2_Q_B, CM2_Q_W, CM2_SK_B, CM2_SK_W, CM2_V_B, CM2_V_W,
  EE0_B, EE0_W, EE1_B, EE1_W, EE2_B, EE2_W,
  FC0_B, FC0_W, FC1_B, FC1_W, FC2_B, FC2_W, FC3_B, FC3_W,
  NE0_B, NE0_W, NE1_B, NE1_W,
  SK2_B, SK2_W, SK3_B, SK3_W,
  N_PARAMS  // == 101
};

// ---------------------------------------------------------------------
// WMMA fragment loader: 16 halves for one lane = two 16B LDS loads
// (K 0..7 at +0, K 16..23 at +16 within the lane's K-base window).
// ---------------------------------------------------------------------
__device__ __forceinline__ v16h frag_ld(const _Float16* p) {
  v8h lo = *(const v8h*)(p);
  v8h hi = *(const v8h*)(p + 16);
  v16h r;
#pragma unroll
  for (int i = 0; i < 8; ++i) { r[i] = lo[i]; r[i + 8] = hi[i]; }
  return r;
}

#define ALD 40  // LDS row pitch in halves (32 + 8 pad -> 16B aligned sub-rows)

// ---------------------------------------------------------------------
// Tiled WMMA GEMM: C[M x NOUT] = act(A[M x K] @ W[K x NOUT] + bias)
// Wt is the f16 weight pre-transposed to [NOUT x K].
// Block = 256 thr = 8 waves; tile M=128; each wave owns a 16-row strip.
// act: 0 = none, 1 = relu, 2 = leaky-relu(0.2)
// ---------------------------------------------------------------------
template <typename TA, typename TC, int K, int NOUT>
__global__ __launch_bounds__(256) void gemm_wmma_kernel(
    const TA* __restrict__ A, const _Float16* __restrict__ Wt,
    const float* __restrict__ bias, TC* __restrict__ C, int M, int act) {
  __shared__ __align__(16) _Float16 As[128 * ALD];
  __shared__ __align__(16) _Float16 Bs[NOUT * ALD];
  const int t = threadIdx.x;
  const int wave = t >> 5;
  const int lane = t & 31;
  const int m0 = blockIdx.x * 128;
  constexpr int NT = NOUT / 16;

  v8f acc[NT];
#pragma unroll
  for (int i = 0; i < NT; ++i) { v8f z = {}; acc[i] = z; }

  for (int k0 = 0; k0 < K; k0 += 32) {
    // Stage A tile (128 x 32) f32/f16 -> f16 LDS
#pragma unroll
    for (int i = 0; i < 16; ++i) {
      int idx = i * 256 + t;
      int r = idx >> 5, c = idx & 31;
      float v = 0.0f;
      int row = m0 + r;
      if (row < M) v = (float)A[(long long)row * K + (k0 + c)];
      As[r * ALD + c] = (_Float16)v;
    }
    // Stage B tile (NOUT x 32) from pre-transposed f16 weights
#pragma unroll
    for (int i = 0; i < (NOUT * 32) / 256; ++i) {
      int idx = i * 256 + t;
      int n = idx >> 5, c = idx & 31;
      Bs[n * ALD + c] = Wt[n * K + (k0 + c)];
    }
    __syncthreads();

    const int kb = (lane >> 4) * 8;          // K base: 0 or 8 per half-wave
    const int am = wave * 16 + (lane & 15);  // A row within tile
    v16h a = frag_ld(&As[am * ALD + kb]);
#pragma unroll
    for (int nt = 0; nt < NT; ++nt) {
      const int bn = nt * 16 + (lane & 15);  // B column (column-major frag)
      v16h b = frag_ld(&Bs[bn * ALD + kb]);
      acc[nt] = __builtin_amdgcn_wmma_f32_16x16x32_f16(
          false, a, false, b, (short)0, acc[nt], false, false);
    }
    __syncthreads();
  }

  // Epilogue: D layout -> lanes 0-15: M=r, N=lane; lanes 16-31: M=8+r, N=lane-16
  const int mbase = m0 + wave * 16 + ((lane >> 4) * 8);
  const int ncol = lane & 15;
#pragma unroll
  for (int nt = 0; nt < NT; ++nt) {
    const int col = nt * 16 + ncol;
    const float bv = bias ? bias[col] : 0.0f;
#pragma unroll
    for (int r = 0; r < 8; ++r) {
      const int row = mbase + r;
      if (row < M) {
        float v = acc[nt][r] + bv;
        if (act == 1) v = v > 0.0f ? v : 0.0f;
        else if (act == 2) v = v > 0.0f ? v : 0.2f * v;
        C[(long long)row * NOUT + col] = (TC)v;
      }
    }
  }
}

// ---------------------------------------------------------------------
// Small utility kernels
// ---------------------------------------------------------------------
__global__ void wconv_kernel(const float* __restrict__ w, _Float16* __restrict__ wt,
                             int K, int N) {
  int i = blockIdx.x * blockDim.x + threadIdx.x;
  if (i >= K * N) return;
  int k = i / N, n = i - k * N;
  wt[n * K + k] = (_Float16)w[i];
}

__global__ void zero_kernel(float* __restrict__ p, int n) {
  int i = blockIdx.x * blockDim.x + threadIdx.x;
  if (i < n) p[i] = 0.0f;
}

__global__ void neginf_kernel(float* __restrict__ p, int n) {
  int i = blockIdx.x * blockDim.x + threadIdx.x;
  if (i < n) p[i] = __int_as_float(0xFF800000);  // -inf
}

// GINE message + scatter-add:  agg[dst] += relu(x[src] + eL)
__global__ void gine_scatter_kernel(const float* __restrict__ x,
                                    const float* __restrict__ eL,
                                    const int* __restrict__ src,
                                    const int* __restrict__ dst,
                                    float* __restrict__ agg, int E) {
  int i = blockIdx.x * blockDim.x + threadIdx.x;
  if (i >= E * HID) return;
  int e = i >> 7, c = i & 127;
  float m = x[src[e] * HID + c] + eL[i];
  m = m > 0.0f ? m : 0.0f;
  atomicAdd(&agg[dst[e] * HID + c], m);
}

// t = (1+eps)*x + agg
__global__ void gine_combine_kernel(const float* __restrict__ x,
                                    const float* __restrict__ agg,
                                    const float* __restrict__ eps,
                                    float* __restrict__ out, int n) {
  int i = blockIdx.x * blockDim.x + threadIdx.x;
  if (i >= n) return;
  out[i] = (1.0f + eps[0]) * x[i] + agg[i];
}

// LayerNorm over HID; pre = A (+B) (+C2); optional leaky-relu on pre.
__global__ __launch_bounds__(128) void ln_kernel(
    const float* __restrict__ A, const float* __restrict__ B,
    const float* __restrict__ C2, const float* __restrict__ gam,
    const float* __restrict__ bet, float* __restrict__ out,
    float* __restrict__ out2, int dolrelu) {
  const int row = blockIdx.x, t = threadIdx.x;
  float v = A[row * HID + t];
  if (B)  v += B[row * HID + t];
  if (C2) v += C2[row * HID + t];
  if (dolrelu) v = v > 0.0f ? v : 0.2f * v;
  __shared__ float red[HID];
  red[t] = v; __syncthreads();
  for (int s = 64; s > 0; s >>= 1) { if (t < s) red[t] += red[t + s]; __syncthreads(); }
  float mu = red[0] * (1.0f / HID); __syncthreads();
  float d = v - mu;
  red[t] = d * d; __syncthreads();
  for (int s = 64; s > 0; s >>= 1) { if (t < s) red[t] += red[t + s]; __syncthreads(); }
  float var = red[0] * (1.0f / HID);
  float r = d * rsqrtf(var + 1e-5f) * gam[t] + bet[t];
  out[row * HID + t] = r;
  if (out2) out2[row * HID + t] = r;
}

// BatchNorm stats: per-channel sum / sumsq of pre = A (+B), optional lrelu
__global__ __launch_bounds__(128) void bn_stats_kernel(
    const float* __restrict__ A, const float* __restrict__ B, int N,
    int dolrelu, float* __restrict__ stats) {
  const int c = threadIdx.x;
  float s = 0.0f, s2 = 0.0f;
  for (int r = blockIdx.x; r < N; r += gridDim.x) {
    float v = A[r * HID + c];
    if (B) v += B[r * HID + c];
    if (dolrelu) v = v > 0.0f ? v : 0.2f * v;
    s += v; s2 += v * v;
  }
  atomicAdd(&stats[c], s);
  atomicAdd(&stats[HID + c], s2);
}

__global__ void bn_apply_kernel(const float* __restrict__ A,
                                const float* __restrict__ B,
                                const float* __restrict__ stats,
                                const float* __restrict__ gam,
                                const float* __restrict__ bet,
                                float* __restrict__ out, float* __restrict__ out2,
                                int N, int dolrelu) {
  int i = blockIdx.x * blockDim.x + threadIdx.x;
  if (i >= N * HID) return;
  int c = i & 127;
  float inv = 1.0f / (float)N;
  float mu = stats[c] * inv;
  float var = stats[HID + c] * inv - mu * mu;
  float v = A[i];
  if (B) v += B[i];
  if (dolrelu) v = v > 0.0f ? v : 0.2f * v;
  float r = (v - mu) * rsqrtf(var + 1e-5f) * gam[c] + bet[c];
  out[i] = r;
  if (out2) out2[i] = r;
}

// float atomic max via monotone int punning (init value must be -inf)
__device__ __forceinline__ void atomicMaxF(float* addr, float val) {
  if (val >= 0.0f) atomicMax((int*)addr, __float_as_int(val));
  else             atomicMin((unsigned int*)addr, __float_as_uint(val));
}

// alpha[e,h] = dot(q[dst], k[src]+ee) / 4 ; amax[dst,h] = max(...)
__global__ void attn_alpha_kernel(const float* __restrict__ q,
                                  const float* __restrict__ k,
                                  const float* __restrict__ ee,
                                  const int* __restrict__ src,
                                  const int* __restrict__ dst,
                                  float* __restrict__ alpha,
                                  float* __restrict__ amax, int E) {
  int i = blockIdx.x * blockDim.x + threadIdx.x;
  if (i >= E * 8) return;
  int e = i >> 3, h = i & 7;
  int s = src[e], d = dst[e];
  const float* qp = q + d * HID + h * 16;
  const float* kp = k + s * HID + h * 16;
  const float* ep = ee + (long long)e * HID + h * 16;
  float acc = 0.0f;
#pragma unroll
  for (int j = 0; j < 16; ++j) acc += qp[j] * (kp[j] + ep[j]);
  acc *= 0.25f;  // 1/sqrt(16)
  alpha[i] = acc;
  atomicMaxF(&amax[d * 8 + h], acc);
}

// ea = exp(alpha - fix(amax[dst])) ; den[dst] += ea   (alpha -> ea in place)
__global__ void attn_ea_kernel(float* __restrict__ alpha,
                               const float* __restrict__ amax,
                               const int* __restrict__ dst,
                               float* __restrict__ den, int E) {
  int i = blockIdx.x * blockDim.x + threadIdx.x;
  if (i >= E * 8) return;
  int e = i >> 3, h = i & 7;
  int d = dst[e];
  float m = amax[d * 8 + h];
  if (m < -1.0e37f) m = 0.0f;  // nodes with no incoming edges (-inf) -> 0
  float ea = expf(alpha[i] - m);
  alpha[i] = ea;
  atomicAdd(&den[d * 8 + h], ea);
}

// out[dst] += (v[src]+ee) * ea/(den[dst]+1e-16)
__global__ void attn_msg_kernel(const float* __restrict__ v,
                                const float* __restrict__ ee,
                                const float* __restrict__ ea,
                                const float* __restrict__ den,
                                const int* __restrict__ src,
                                const int* __restrict__ dst,
                                float* __restrict__ out, int E) {
  int i = blockIdx.x * blockDim.x + threadIdx.x;
  if (i >= E * HID) return;
  int e = i >> 7, c = i & 127, h = c >> 4;
  int s = src[e], d = dst[e];
  float a = ea[e * 8 + h] / (den[d * 8 + h] + 1e-16f);
  float m = (v[s * HID + c] + ee[i]) * a;
  atomicAdd(&out[d * HID + c], m);
}

// global mean pool accumulation
__global__ void pool_kernel(const float* __restrict__ h,
                            const int* __restrict__ batch,
                            float* __restrict__ gsum, float* __restrict__ gcnt,
                            int N) {
  int i = blockIdx.x * blockDim.x + threadIdx.x;
  if (i >= N * HID) return;
  int r = i >> 7, c = i & 127;
  int b = batch[r];
  atomicAdd(&gsum[b * HID + c], h[i]);
  if (c == 0) atomicAdd(&gcnt[b], 1.0f);
}

// FC head: tiny, one block per graph
__global__ __launch_bounds__(256) void fc_head_kernel(
    const float* __restrict__ gsum, const float* __restrict__ gcnt,
    const float* __restrict__ w0, const float* __restrict__ b0,
    const float* __restrict__ w1, const float* __restrict__ b1,
    const float* __restrict__ w2, const float* __restrict__ b2,
    const float* __restrict__ w3, const float* __restrict__ b3,
    float* __restrict__ out) {
  const int gid = blockIdx.x, t = threadIdx.x;
  __shared__ float g[128], h1[256], h2[12], h3[12];
  if (t < 128) {
    float c = gcnt[gid]; c = c < 1.0f ? 1.0f : c;
    g[t] = gsum[gid * 128 + t] / c;
  }
  __syncthreads();
  { float a = b0[t];
    for (int k = 0; k < 128; ++k) a += g[k] * w0[k * 256 + t];
    h1[t] = a > 0.0f ? a : 0.0f; }
  __syncthreads();
  if (t < 12) { float a = b1[t];
    for (int k = 0; k < 256; ++k) a += h1[k] * w1[k * 12 + t];
    h2[t] = a > 0.0f ? a : 0.0f; }
  __syncthreads();
  if (t < 12) { float a = b2[t];
    for (int k = 0; k < 12; ++k) a += h2[k] * w2[k * 12 + t];
    h3[t] = a > 0.0f ? a : 0.0f; }
  __syncthreads();
  if (t < 6) { float a = b3[t];
    for (int k = 0; k < 12; ++k) a += h3[k] * w3[k * 6 + t];
    out[gid * 6 + t] = a; }
}

// ---------------------------------------------------------------------
// Host orchestration
// ---------------------------------------------------------------------
static inline int cdiv(int a, int b) { return (a + b - 1) / b; }

template <typename TA, typename TC, int K, int NOUT>
static inline void launch_gemm(const TA* A, const _Float16* Wt, const float* b,
                               TC* C, int M, int act, hipStream_t s) {
  gemm_wmma_kernel<TA, TC, K, NOUT><<<cdiv(M, 128), 256, 0, s>>>(A, Wt, b, C, M, act);
}

extern "C" void kernel_launch(void* const* d_in, const int* in_sizes, int n_in,
                              void* d_out, int out_size, void* d_ws, size_t ws_size,
                              hipStream_t stream) {
  (void)in_sizes; (void)out_size; (void)ws_size;
  if (n_in < N_PARAMS) return;

  const int N = NNODES, E = NEDGES, G = NGRAPH;
  const float* x       = (const float*)d_in[IN_X];
  const int*   eidx    = (const int*)d_in[IN_EIDX];
  const int*   src     = eidx;
  const int*   dst     = eidx + E;
  const float* eattr   = (const float*)d_in[IN_EATTR];
  const int*   batch   = (const int*)d_in[IN_BATCH];
  auto P = [&](int i) { return (const float*)d_in[i]; };

  // ----- workspace bump allocator -----
  char* wp = (char*)d_ws;
  auto alloc = [&](size_t bytes) -> void* {
    void* p = (void*)wp;
    wp += (bytes + 255) & ~(size_t)255;
    return p;
  };
  _Float16* e16   = (_Float16*)alloc((size_t)E * HID * 2);   // 153.6 MB
  float*    eL    = (float*)alloc((size_t)E * HID * 4);      // 307.2 MB (also chunk scratch)
  float*    alphaB= (float*)alloc((size_t)E * 8 * 4);        // 19.2 MB
  float*    amax  = (float*)alloc((size_t)N * 8 * 4);
  float*    den   = (float*)alloc((size_t)N * 8 * 4);
  float* nb[9];
  for (int i = 0; i < 9; ++i) nb[i] = (float*)alloc((size_t)N * HID * 4);
  float*    stats = (float*)alloc(2 * HID * 4);
  float*    gsum  = (float*)alloc((size_t)G * HID * 4);
  float*    gcnt  = (float*)alloc((size_t)G * 4);

  auto zero = [&](float* p, int n) {
    zero_kernel<<<cdiv(n, 256), 256, 0, stream>>>(p, n);
  };
  // ----- weight conversion (f32 [K][N] -> f16 [N][K]) -----
  auto wconv = [&](int widx, int K, int Nn) -> _Float16* {
    _Float16* wt = (_Float16*)alloc((size_t)K * Nn * 2);
    wconv_kernel<<<cdiv(K * Nn, 256), 256, 0, stream>>>(P(widx), wt, K, Nn);
    return wt;
  };
  _Float16* ne0W = wconv(NE0_W, 64, 128);
  _Float16* ne1W = wconv(NE1_W, 128, 128);
  _Float16* ee0W = wconv(EE0_W, 32, 256);
  _Float16* ee1W = wconv(EE1_W, 256, 256);
  _Float16* ee2W = wconv(EE2_W, 256, 128);
  _Float16 *g2aL = wconv(C2A_LIN_W,128,128), *g2a1 = wconv(C2A_NN1_W,128,128), *g2a2 = wconv(C2A_NN2_W,128,128);
  _Float16 *g2bL = wconv(C2B_LIN_W,128,128), *g2b1 = wconv(C2B_NN1_W,128,128), *g2b2 = wconv(C2B_NN2_W,128,128);
  _Float16 *g3aL = wconv(C3A_LIN_W,128,128), *g3a1 = wconv(C3A_NN1_W,128,128), *g3a2 = wconv(C3A_NN2_W,128,128);
  _Float16 *g3bL = wconv(C3B_LIN_W,128,128), *g3b1 = wconv(C3B_NN1_W,128,128), *g3b2 = wconv(C3B_NN2_W,128,128);
  _Float16 *m2E = wconv(CM2_E_W,128,128), *m2K = wconv(CM2_K_W,128,128), *m2Q = wconv(CM2_Q_W,128,128),
           *m2S = wconv(CM2_SK_W,128,128), *m2V = wconv(CM2_V_W,128,128);
  _Float16 *mE  = wconv(CM_E_W,128,128),  *mK  = wconv(CM_K_W,128,128),  *mQ  = wconv(CM_Q_W,128,128),
           *mS  = wconv(CM_SK_W,128,128), *mV  = wconv(CM_V_W,128,128);
  _Float16 *fE  = wconv(CF_E_W,128,128),  *fK  = wconv(CF_K_W,128,128),  *fQ  = wconv(CF_Q_W,128,128),
           *fS  = wconv(CF_SK_W,128,128), *fV  = wconv(CF_V_W,128,128);
  _Float16* sk2W = wconv(SK2_W, 128, 128);
  _Float16* sk3W = wconv(SK3_W, 128, 128);

  // ----- edge encoder: Lin(32->256)+ReLU, Lin(256->256)+ReLU, Lin(256->128)
  // chunked; both 256-wide intermediates overlay the (not-yet-used) eL buffer
  const int CH = 150000;
  float* ec0 = eL;
  float* ec1 = eL + (size_t)CH * 256;
  for (int c0 = 0; c0 < E; c0 += CH) {
    int len = (E - c0 < CH) ? (E - c0) : CH;
    launch_gemm<float, float, 32, 256>(eattr + (size_t)c0 * 32, ee0W, P(EE0_B), ec0, len, 1, stream);
    launch_gemm<float, float, 256, 256>(ec0, ee1W, P(EE1_B), ec1, len, 1, stream);
    launch_gemm<float, _Float16, 256, 128>(ec1, ee2W, P(EE2_B), e16 + (size_t)c0 * HID, len, 0, stream);
  }

  float *h = nb[0], *xinit = nb[1], *xfirst = nb[2], *xsecond = nb[3];
  float *t0 = nb[4], *t1 = nb[5], *t2 = nb[6], *t3 = nb[7], *agg = nb[8];

  // ----- node encoder -----
  launch_gemm<float, float, 64, 128>(x, ne0W, P(NE0_B), t0, N, 1, stream);
  launch_gemm<float, float, 128, 128>(t0, ne1W, P(NE1_B), xinit, N, 0, stream);

  // ----- GINE conv: hout = LN(lrelu(gine(hin))) -----
  auto run_gine = [&](const float* hin, _Float16* linW, int linB, int eps,
                      _Float16* n1W, int n1B, _Float16* n2W, int n2B,
                      int lnG, int lnB, float* hout) {
    launch_gemm<_Float16, float, 128, 128>(e16, linW, P(linB), eL, E, 0, stream);
    zero(agg, N * HID);
    gine_scatter_kernel<<<cdiv(E * HID, 256), 256, 0, stream>>>(hin, eL, src, dst, agg, E);
    gine_combine_kernel<<<cdiv(N * HID, 256), 256, 0, stream>>>(hin, agg, P(eps), t0, N * HID);
    launch_gemm<float, float, 128, 128>(t0, n1W, P(n1B), t1, N, 1, stream);
    launch_gemm<float, float, 128, 128>(t1, n2W, P(n2B), t0, N, 1, stream);
    ln_kernel<<<N, 128, 0, stream>>>(t0, nullptr, nullptr, P(lnG), P(lnB), hout, nullptr, 1);
  };

  // ----- TransformerConv: leaves attn-out in t3, skip in agg -----
  auto run_tconv = [&](const float* hin, _Float16* qW, int qB, _Float16* kW, int kB,
                       _Float16* vW, int vB, _Float16* eW, _Float16* skW, int skB) {
    launch_gemm<float, float, 128, 128>(hin, qW, P(qB), t0, N, 0, stream);
    launch_gemm<float, float, 128, 128>(hin, kW, P(kB), t1, N, 0, stream);
    launch_gemm<float, float, 128, 128>(hin, vW, P(vB), t2, N, 0, stream);
    launch_gemm<_Float16, float, 128, 128>(e16, eW, nullptr, eL, E, 0, stream);  // ee
    neginf_kernel<<<cdiv(N * 8, 256), 256, 0, stream>>>(amax, N * 8);
    zero(den, N * 8);
    attn_alpha_kernel<<<cdiv(E * 8, 256), 256, 0, stream>>>(t0, t1, eL, src, dst, alphaB, amax, E);
    attn_ea_kernel<<<cdiv(E * 8, 256), 256, 0, stream>>>(alphaB, amax, dst, den, E);
    zero(t3, N * HID);
    attn_msg_kernel<<<cdiv(E * HID, 256), 256, 0, stream>>>(t2, eL, alphaB, den, src, dst, t3, E);
    launch_gemm<float, float, 128, 128>(hin, skW, P(skB), agg, N, 0, stream);
  };

  // ===== block 2 =====
  run_gine(xinit, g2aL, C2A_LIN_B, C2A_EPS, g2a1, C2A_NN1_B, g2a2, C2A_NN2_B, BN2A_G, BN2A_B, h);
  run_gine(h,     g2bL, C2B_LIN_B, C2B_EPS, g2b1, C2B_NN1_B, g2b2, C2B_NN2_B, BN2B_G, BN2B_B, h);
  launch_gemm<float, float, 128, 128>(xinit, sk2W, P(SK2_B), t1, N, 0, stream);
  zero(stats, 2 * HID);
  bn_stats_kernel<<<512, 128, 0, stream>>>(h, t1, N, 0, stats);
  bn_apply_kernel<<<cdiv(N * HID, 256), 256, 0, stream>>>(h, t1, stats, P(BN2BLK_G), P(BN2BLK_B), h, xfirst, N, 0);

  // ===== conv_mid2 =====
  run_tconv(h, m2Q, CM2_Q_B, m2K, CM2_K_B, m2V, CM2_V_B, m2E, m2S, CM2_SK_B);
  ln_kernel<<<N, 128, 0, stream>>>(t3, agg, nullptr, P(BNMID2_G), P(BNMID2_B), h, xinit, 1);

  // ===== block 3 =====
  run_gine(h, g3aL, C3A_LIN_B, C3A_EPS, g3a1, C3A_NN1_B, g3a2, C3A_NN2_B, BN3A_G, BN3A_B, h);
  run_gine(h, g3bL, C3B_LIN_B, C3B_EPS, g3b1, C3B_NN1_B, g3b2, C3B_NN2_B, BN3B_G, BN3B_B, h);
  launch_gemm<float, float, 128, 128>(xinit, sk3W, P(SK3_B), t1, N, 0, stream);
  zero(stats, 2 * HID);
  bn_stats_kernel<<<512, 128, 0, stream>>>(h, t1, N, 0, stats);
  bn_apply_kernel<<<cdiv(N * HID, 256), 256, 0, stream>>>(h, t1, stats, P(BN3BLK_G), P(BN3BLK_B), h, xsecond, N, 0);

  // ===== conv_mid =====
  run_tconv(h, mQ, CM_Q_B, mK, CM_K_B, mV, CM_V_B, mE, mS, CM_SK_B);
  ln_kernel<<<N, 128, 0, stream>>>(t3, agg, nullptr, P(BNMID_G), P(BNMID_B), h, nullptr, 1);

  // h = LN(h + x_second + x_first)
  ln_kernel<<<N, 128, 0, stream>>>(h, xsecond, xfirst, P(BNMID3_G), P(BNMID3_B), h, nullptr, 0);

  // ===== conv_final + BN(lrelu(.)) =====
  run_tconv(h, fQ, CF_Q_B, fK, CF_K_B, fV, CF_V_B, fE, fS, CF_SK_B);
  zero(stats, 2 * HID);
  bn_stats_kernel<<<512, 128, 0, stream>>>(t3, agg, N, 1, stats);
  bn_apply_kernel<<<cdiv(N * HID, 256), 256, 0, stream>>>(t3, agg, stats, P(BNFIN_G), P(BNFIN_B), h, nullptr, N, 1);

  // ===== pool + FC head =====
  zero(gsum, G * HID);
  zero(gcnt, G);
  pool_kernel<<<cdiv(N * HID, 256), 256, 0, stream>>>(h, batch, gsum, gcnt, N);
  fc_head_kernel<<<G, 256, 0, stream>>>(gsum, gcnt,
      P(FC0_W), P(FC0_B), P(FC1_W), P(FC1_B),
      P(FC2_W), P(FC2_B), P(FC3_W), P(FC3_B), (float*)d_out);
}
